// LSTMLayer_7112465842327
// MI455X (gfx1250) — matrix operand order repaired and checked by reading, not compiled, not run on hardware
//
#include <hip/hip_runtime.h>

typedef __attribute__((ext_vector_type(8)))  float  v8f;
typedef __attribute__((ext_vector_type(16))) __bf16 v16bf;
typedef __attribute__((ext_vector_type(8)))  __bf16 v8bf;
typedef __attribute__((ext_vector_type(4)))  int    v4i;

#define B_SZ   8192
#define D_K    1024
#define K2     2048              // fused K (x-part then h-part)
#define NN     (1024*1024)       // elements per weight matrix
#define HC     ((size_t)8192*1024)
#define LDH    40                // padded LDS row stride in halves (80 bytes)
#define ABUF   (128*LDH)         // halves per A buffer
#define BBUF   (4*32*LDH)        // halves per B buffer

#if defined(__gfx1250__) && __has_builtin(__builtin_amdgcn_global_load_async_to_lds_b128) && \
    __has_builtin(__builtin_amdgcn_s_wait_asynccnt)
#define HAS_ASYNC 1
#else
#define HAS_ASYNC 0
#endif

#if HAS_ASYNC
__device__ __forceinline__ void async_cp16(const __bf16* g, __bf16* l) {
    __builtin_amdgcn_global_load_async_to_lds_b128(
        (__attribute__((address_space(1))) v4i*)(v4i*)(void*)g,
        (__attribute__((address_space(3))) v4i*)(v4i*)(void*)l, 0, 0);
}
#endif

// ------- f32 -> bf16 convert into strided dst (xh interleave), 8 elems/thread -------
__global__ void cvt_bf16_k(const float* __restrict__ src, __bf16* __restrict__ dst,
                           int dstStride) {
    int i   = blockIdx.x * blockDim.x + threadIdx.x;   // one per 8 elements
    int row = i >> 7;                                  // 128 threads per 1024-col row
    int col = (i & 127) * 8;
    const float4* s = (const float4*)(src + (size_t)row * D_K + col);
    float4 f0 = s[0], f1 = s[1];
    v8bf o;
    o[0]=(__bf16)f0.x; o[1]=(__bf16)f0.y; o[2]=(__bf16)f0.z; o[3]=(__bf16)f0.w;
    o[4]=(__bf16)f1.x; o[5]=(__bf16)f1.y; o[6]=(__bf16)f1.z; o[7]=(__bf16)f1.w;
    *(v8bf*)(dst + (size_t)row * dstStride + col) = o;
}

// ------------- W[N,K] f32 -> Wt[K,N] bf16, 32x32 LDS tile transpose -------------
__global__ void transp_bf16_k(const float* __restrict__ W, __bf16* __restrict__ Wt) {
    __shared__ float tile[32][33];
    const int bx = blockIdx.x;          // k-tile
    const int by = blockIdx.y;          // n-tile
    const int tx = threadIdx.x & 31;
    const int ty = threadIdx.x >> 5;    // 0..7
#pragma unroll
    for (int i = 0; i < 4; ++i) {
        int n = by*32 + ty + i*8;
        int k = bx*32 + tx;
        tile[ty + i*8][tx] = W[(size_t)n*D_K + k];
    }
    __syncthreads();
#pragma unroll
    for (int i = 0; i < 4; ++i) {
        int k = bx*32 + ty + i*8;
        int n = by*32 + tx;
        Wt[(size_t)k*D_K + n] = (__bf16)tile[tx][ty + i*8];
    }
}

// ---------------- bias pair sum ----------------
__global__ void bias_sum_k(const float* __restrict__ a, const float* __restrict__ b,
                           float* __restrict__ o) {
    int i = blockIdx.x * blockDim.x + threadIdx.x;
    if (i < D_K) o[i] = a[i] + b[i];
}

// ---------------- fused 8-GEMM + LSTM pointwise ----------------
// grid (64, 32), block 256.  Block tile: M=128, N=32 (per gate), K=2048 fused.
// xh: [8192][2048] bf16 (x | h).  Wt: [gate: i,f,g,o][2048][1024] bf16 (i2h rows then h2h rows).
__global__ __launch_bounds__(256)
void lstm_wmma_k(const __bf16* __restrict__ xh,  const __bf16* __restrict__ Wt,
                 const float* __restrict__ bsum, const float* __restrict__ c0,
                 float* __restrict__ outBuf) {
    __shared__ __align__(16) __bf16 sA[3 * ABUF];     // 30 KB
    __shared__ __align__(16) __bf16 sB[3 * BBUF];     // 30 KB

    const int t    = threadIdx.x;
    const int bM   = blockIdx.x;
    const int bN   = blockIdx.y;
    const int lane = t & 31;
    const int lo   = lane & 15;
    const int hi   = lane >> 4;
    const int wave = t >> 5;
    const int wm   = wave & 3;
    const int wn   = wave >> 2;

    // global->LDS staging assignment
    const int lr = t >> 2;            // 0..63 (A rows; also +64)
    const int lc = (t & 3) * 8;       // 0,8,16,24 (k within 32-wide slab)
    const int g0 = t >> 7;            // 0/1 ; second chunk gate = g0+2
    const int lk = (t >> 2) & 31;     // 0..31 (B k-row)

    // per-thread linear base pointers (advance by kt*32 along fused K)
    const __bf16* aG0 = xh + (size_t)(bM*128 + lr)      * K2 + lc;
    const __bf16* aG1 = xh + (size_t)(bM*128 + lr + 64) * K2 + lc;
    const __bf16* bG0 = Wt + (size_t)(2*g0)     * NN + (size_t)lk * D_K + bN*32 + lc;
    const __bf16* bG1 = Wt + (size_t)(2*g0 + 4) * NN + (size_t)lk * D_K + bN*32 + lc;

    v8f acc[4][2] = {};

    // WMMA compute for one LDS buffer: 2 A-frags, 4 B-frags, 8 WMMAs
    auto compute = [&](int bufI) {
        const __bf16* a = sA + bufI * ABUF;
        const __bf16* b = sB + bufI * BBUF;
        union U { uint4 u[2]; v16bf v; } u;
        v16bf af[2], bfr[4];
        // A frag: lanes 0-15 -> M=lo, K {0-7,16-23}; lanes 16-31 -> K {8-15,24-31}
        const __bf16* aw = a + (wm*32 + lo) * LDH + hi * 8;
        u.u[0] = *(const uint4*)(aw);
        u.u[1] = *(const uint4*)(aw + 16);
        af[0] = u.v;
        u.u[0] = *(const uint4*)(aw + 16*LDH);
        u.u[1] = *(const uint4*)(aw + 16*LDH + 16);
        af[1] = u.v;
        // B frags: lane -> N=lo column; lanes 0-15 K {0-7,16-23}, lanes 16-31 K {8-15,24-31}
#pragma unroll
        for (int g = 0; g < 4; ++g) {
            const __bf16* bp = b + (g*32 + hi*8) * LDH + wn*16 + lo;
#pragma unroll
            for (int s = 0; s < 8; ++s) {
                bfr[g][s]     = bp[s * LDH];
                bfr[g][s + 8] = bp[(16 + s) * LDH];
            }
        }
#pragma unroll
        for (int g = 0; g < 4; ++g)
#pragma unroll
            for (int mt = 0; mt < 2; ++mt)
                acc[g][mt] = __builtin_amdgcn_wmma_f32_16x16x32_bf16(
                    false, af[mt], false, bfr[g], (short)0, acc[g][mt], false, false);
    };

#if HAS_ASYNC
    // -------- async direct-to-LDS, 3 buffers, 2 tiles in flight, 1 barrier/iter --------
    auto issue = [&](int kt, int bufI) {
        const int kk = kt * 32;                 // fused-K offset (halves)
        __bf16* a = sA + bufI * ABUF;
        __bf16* b = sB + bufI * BBUF;
        async_cp16(aG0 + kk,                a + lr        * LDH + lc);
        async_cp16(aG1 + kk,                b == b ? a + (lr + 64) * LDH + lc
                                                  : a);                 // (kept simple below)
        async_cp16(bG0 + (size_t)kk * D_K,  b + (g0*32     + lk) * LDH + lc);
        async_cp16(bG1 + (size_t)kk * D_K,  b + ((g0+2)*32 + lk) * LDH + lc);
    };
    issue(0, 0);
    issue(1, 1);
    for (int kt = 0; kt < 60; kt += 3) {
        __builtin_amdgcn_s_wait_asynccnt(4); __syncthreads();
        compute(0); issue(kt + 2, 2);
        __builtin_amdgcn_s_wait_asynccnt(4); __syncthreads();
        compute(1); issue(kt + 3, 0);
        __builtin_amdgcn_s_wait_asynccnt(4); __syncthreads();
        compute(2); issue(kt + 4, 1);
    }
    // drain: tiles 60(b0), 61(b1), 62(b2), 63(b0)
    __builtin_amdgcn_s_wait_asynccnt(4); __syncthreads();
    compute(0); issue(62, 2);
    __builtin_amdgcn_s_wait_asynccnt(4); __syncthreads();
    compute(1); issue(63, 0);
    __builtin_amdgcn_s_wait_asynccnt(4); __syncthreads();
    compute(2);
    __builtin_amdgcn_s_wait_asynccnt(0); __syncthreads();
    compute(0);
#else
    // -------- fallback: VGPR-staged double buffer, one barrier/iter --------
    uint4 ra0, ra1, rb0, rb1;
    auto gload = [&](int kt) {
        const int kk = kt * 32;
        ra0 = *(const uint4*)(aG0 + kk);
        ra1 = *(const uint4*)(aG1 + kk);
        rb0 = *(const uint4*)(bG0 + (size_t)kk * D_K);
        rb1 = *(const uint4*)(bG1 + (size_t)kk * D_K);
    };
    gload(0);
    int buf = 0;
    for (int kt = 0; kt < 64; ++kt) {
        __bf16* a = sA + buf * ABUF;
        __bf16* b = sB + buf * BBUF;
        *(uint4*)(a + lr        * LDH + lc) = ra0;
        *(uint4*)(a + (lr + 64) * LDH + lc) = ra1;
        *(uint4*)(b + (g0*32     + lk) * LDH + lc) = rb0;
        *(uint4*)(b + ((g0+2)*32 + lk) * LDH + lc) = rb1;
        __syncthreads();
        if (kt + 1 < 64) gload(kt + 1);
        compute(buf);
        buf ^= 1;
    }
#endif

    // -------- LSTM pointwise epilogue, straight from accumulators --------
    const int   ng  = bN*32 + wn*16 + lo;
    const float bs0 = bsum[ng];
    const float bs1 = bsum[1024 + ng];
    const float bs2 = bsum[2048 + ng];
    const float bs3 = bsum[3072 + ng];
    const int   mB  = bM*128 + wm*32 + hi*8;
    float* ctP = outBuf + HC;

#pragma unroll
    for (int mt = 0; mt < 2; ++mt)
#pragma unroll
        for (int r = 0; r < 8; ++r) {
            const int    m   = mB + mt*16 + r;
            const size_t off = (size_t)m * D_K + ng;
            const float zi = acc[0][mt][r] + bs0;
            const float zf = acc[1][mt][r] + bs1;
            const float zg = acc[2][mt][r] + bs2;
            const float zo = acc[3][mt][r] + bs3;
            const float it = 1.f / (1.f + __expf(-zi));
            const float ft = 1.f / (1.f + __expf(-zf));
            const float ot = 1.f / (1.f + __expf(-zo));
            const float gt = 2.f / (1.f + __expf(-2.f*zg)) - 1.f;
            const float cn = ft * c0[off] + it * gt;
            const float th = 2.f / (1.f + __expf(-2.f*cn)) - 1.f;
            outBuf[off] = ot * th;
            ctP[off]    = cn;
        }
}

extern "C" void kernel_launch(void* const* d_in, const int* in_sizes, int n_in,
                              void* d_out, int out_size, void* d_ws, size_t ws_size,
                              hipStream_t stream) {
    const float* x  = (const float*)d_in[0];
    const float* h0 = (const float*)d_in[1];
    const float* c0 = (const float*)d_in[2];

    char* ws = (char*)d_ws;
    __bf16* xh   = (__bf16*)ws;                      // [8192][2048] = 32 MB
    __bf16* Wt   = xh + (size_t)B_SZ * K2;           // 8 x 1M bf16 = 16 MB
    float*  bsum = (float*)(Wt + (size_t)8 * NN);    // 16 KB

    const int n8 = (B_SZ * D_K) / 8;                 // 1,048,576 threads-worth
    cvt_bf16_k<<<n8 / 256, 256, 0, stream>>>(x,  xh,        K2);
    cvt_bf16_k<<<n8 / 256, 256, 0, stream>>>(h0, xh + D_K,  K2);

    // Wt layout: [gate i,f,g,o][i2h 1024 rows | h2h 1024 rows][1024]
    for (int g = 0; g < 4; ++g) {
        transp_bf16_k<<<dim3(32, 32), 256, 0, stream>>>(
            (const float*)d_in[3 + 4*g], Wt + (size_t)(2*g)     * NN);
        transp_bf16_k<<<dim3(32, 32), 256, 0, stream>>>(
            (const float*)d_in[5 + 4*g], Wt + (size_t)(2*g + 1) * NN);
        bias_sum_k<<<4, 256, 0, stream>>>(
            (const float*)d_in[4 + 4*g], (const float*)d_in[6 + 4*g], bsum + g * 1024);
    }

    lstm_wmma_k<<<dim3(64, 32), 256, 0, stream>>>(xh, Wt, bsum, c0, (float*)d_out);
}
